// LocalTransformer_22265110463244
// MI455X (gfx1250) — matrix operand order repaired and checked by reading, not compiled, not run on hardware
//
#include <hip/hip_runtime.h>
#include <cmath>

typedef __bf16 bf16;
typedef __attribute__((ext_vector_type(16))) __bf16 v16bf;
typedef __attribute__((ext_vector_type(8)))  __bf16 v8bf;
typedef __attribute__((ext_vector_type(8)))  float   v8f;

#define DEV __device__ __forceinline__

DEV v8f zero8() {
  v8f z;
#pragma unroll
  for (int i = 0; i < 8; i++) z[i] = 0.0f;
  return z;
}

DEV unsigned int pack2(float a, float b) {
  union { bf16 h[2]; unsigned int u; } cv;
  cv.h[0] = (bf16)a; cv.h[1] = (bf16)b;
  return cv.u;
}

DEV v8f wmma_bf16(v16bf a, v16bf b, v8f c) {
  return __builtin_amdgcn_wmma_f32_16x16x32_bf16(
      false, a, false, b, (short)0, c, false, false);
}

// A-operand (16xK=32 bf16): lane<16 -> M=lane, elems0-15=K0-15
//                            lane>=16 -> M=lane-16, elems0-15=K16-31
DEV v16bf load_fragA(const bf16* base, int stride, int lane) {
  const int half = lane >> 4;
  const int r = lane & 15;
  const bf16* p = base + r * stride;
  v8bf lo = *(const v8bf*)(p + half * 16);
  v8bf hi = *(const v8bf*)(p + half * 16 + 8);
  v16bf out;
#pragma unroll
  for (int i = 0; i < 8; i++) { out[i] = lo[i]; out[8 + i] = hi[i]; }
  return out;
}

// B-operand (K=32 x 16 bf16), from [N][K] LDS tile: lane<16 -> col=lane, K0-15;
// lane>=16 -> col=lane-16, K16-31 (same contiguous k order as load_fragA)
DEV v16bf load_fragB(const bf16* base, int stride, int lane) {
  return *(const v16bf*)(base + (lane & 15) * stride + (lane >> 4) * 16);
}

// ---------------------------------------------------------------------------
// x[n][d] = tok_emb[tokens[n]][d] + pos_emb[n][d]
__global__ __launch_bounds__(256) void embed_kernel(
    const int* __restrict__ tok, const float* __restrict__ te,
    const float* __restrict__ pe, float* __restrict__ x) {
  const int n = blockIdx.x;
  const size_t t = (size_t)tok[n];
  for (int d = threadIdx.x; d < 1024; d += 256)
    x[(size_t)n * 1024 + d] = te[t * 1024 + d] + pe[(size_t)n * 1024 + d];
}

// ---------------------------------------------------------------------------
// dynamic position bias MLP: tab[r][h] for r = 0..1023, h = 0..15
__global__ __launch_bounds__(512) void dpb_kernel(
    const float* __restrict__ w1, const float* __restrict__ b1,
    const float* __restrict__ w2, const float* __restrict__ b2,
    const float* __restrict__ w3, const float* __restrict__ b3,
    float* __restrict__ tab) {
  __shared__ float h1[512];
  __shared__ float h2[512];
  const int r = blockIdx.x, tid = threadIdx.x;
  const float rel = (float)r;
  {
    float z = rel * w1[tid] + b1[tid];
    h1[tid] = z / (1.0f + __expf(-z));           // silu
  }
  __syncthreads();
  {
    float s = b2[tid];
    for (int k = 0; k < 512; k++) s += h1[k] * w2[k * 512 + tid];
    h2[tid] = s / (1.0f + __expf(-s));
  }
  __syncthreads();
  if (tid < 16) {
    float s = b3[tid];
    for (int k = 0; k < 512; k++) s += h2[k] * w3[k * 16 + tid];
    tab[r * 16 + tid] = s;
  }
}

// ---------------------------------------------------------------------------
// LayerNorm over 1024, output bf16
__global__ __launch_bounds__(256) void ln_kernel(
    const float* __restrict__ x, const float* __restrict__ g,
    const float* __restrict__ b, bf16* __restrict__ out) {
  __shared__ float red[8];
  const int n = blockIdx.x, tid = threadIdx.x;
  const float* row = x + (size_t)n * 1024;
  float s = 0.0f;
  for (int d = tid; d < 1024; d += 256) s += row[d];
#pragma unroll
  for (int sh = 16; sh >= 1; sh >>= 1) s += __shfl_xor(s, sh, 32);
  if ((tid & 31) == 0) red[tid >> 5] = s;
  __syncthreads();
  float mean = 0.0f;
#pragma unroll
  for (int i = 0; i < 8; i++) mean += red[i];
  mean *= (1.0f / 1024.0f);
  __syncthreads();
  float v = 0.0f;
  for (int d = tid; d < 1024; d += 256) { float t = row[d] - mean; v += t * t; }
#pragma unroll
  for (int sh = 16; sh >= 1; sh >>= 1) v += __shfl_xor(v, sh, 32);
  if ((tid & 31) == 0) red[tid >> 5] = v;
  __syncthreads();
  float var = 0.0f;
#pragma unroll
  for (int i = 0; i < 8; i++) var += red[i];
  var *= (1.0f / 1024.0f);
  const float rstd = rsqrtf(var + 1e-5f);
  for (int d = tid; d < 1024; d += 256)
    out[(size_t)n * 1024 + d] = (bf16)((row[d] - mean) * rstd * g[d] + b[d]);
}

// ---------------------------------------------------------------------------
// C[MxN] = A[MxK](bf16) * B[KxN](f32 -> bf16); epi: 0 = store, 1 = += resid
// Double-buffered LDS, register-staged prefetch of the next K-tile.
__global__ __launch_bounds__(256) void gemm_bf16(
    const bf16* __restrict__ A, const float* __restrict__ B,
    float* __restrict__ C, const float* __restrict__ resid,
    int M, int N, int K, int epi) {
  __shared__ __align__(64) bf16 lA[2][128 * 32];   // [m][k]
  __shared__ __align__(64) bf16 lBT[2][128 * 32];  // [n][k]
  const int bm = blockIdx.y * 128;
  const int bn = blockIdx.x * 128;
  const int tid = threadIdx.x;
  const int wid = tid >> 5, lane = tid & 31;
  const int wm = (wid >> 2) * 64;   // wave M offset: 0 / 64
  const int wn = (wid & 3) * 32;    // wave N offset: 0..96
  const bool fullN = (bn + 128 <= N);
  v8f acc[4][2];
#pragma unroll
  for (int i = 0; i < 4; i++)
#pragma unroll
    for (int j = 0; j < 2; j++) acc[i][j] = zero8();
  const int ksteps = (K + 31) >> 5;
  const int rowA = tid >> 1, cbA = (tid & 1) * 16;

  // register staging for one K-tile (A: 16 bf16; B: 2 slots x 2 k-rows x float4)
  unsigned int aReg[8];
  float4 bR0[2], bR1[2];

  auto fetch = [&](int k0) {
    const bool fullK = (k0 + 32 <= K);
    if (fullK) {
      const unsigned int* ap =
          (const unsigned int*)(A + (size_t)(bm + rowA) * K + k0 + cbA);
#pragma unroll
      for (int c = 0; c < 8; c++) aReg[c] = ap[c];
    } else {   // K tail (FF2 last step only)
      union { bf16 h[16]; unsigned int u[8]; } at;
      const bf16* ap = A + (size_t)(bm + rowA) * K;
#pragma unroll
      for (int c = 0; c < 16; c++) {
        const int k = k0 + cbA + c;
        at.h[c] = (k < K) ? ap[k] : (bf16)0.0f;
      }
#pragma unroll
      for (int c = 0; c < 8; c++) aReg[c] = at.u[c];
    }
    if (fullK && fullN) {
#pragma unroll
      for (int i = 0; i < 2; i++) {          // 512 slots: (kpair p, 4 cols n4)
        const int idx = tid + i * 256;
        const int n4 = (idx & 31) * 4;
        const int p = idx >> 5;              // 0..15
        bR0[i] = *(const float4*)(B + (size_t)(k0 + 2 * p) * N + bn + n4);
        bR1[i] = *(const float4*)(B + (size_t)(k0 + 2 * p + 1) * N + bn + n4);
      }
    } else {
#pragma unroll
      for (int i = 0; i < 2; i++) {
        const int idx = tid + i * 256;
        const int n4 = (idx & 31) * 4;
        const int p = idx >> 5;
        const int gk0 = k0 + 2 * p, gk1 = gk0 + 1;
        float t[8];
#pragma unroll
        for (int j = 0; j < 4; j++) {
          const int gn = bn + n4 + j;
          t[j]     = (gk0 < K && gn < N) ? B[(size_t)gk0 * N + gn] : 0.0f;
          t[4 + j] = (gk1 < K && gn < N) ? B[(size_t)gk1 * N + gn] : 0.0f;
        }
        bR0[i] = make_float4(t[0], t[1], t[2], t[3]);
        bR1[i] = make_float4(t[4], t[5], t[6], t[7]);
      }
    }
  };

  auto commit = [&](int buf) {
    unsigned int* dst = (unsigned int*)(lA[buf] + rowA * 32 + cbA);
#pragma unroll
    for (int c = 0; c < 8; c++) dst[c] = aReg[c];
#pragma unroll
    for (int i = 0; i < 2; i++) {
      const int idx = tid + i * 256;
      const int n4 = (idx & 31) * 4;
      const int p = idx >> 5;
      bf16* bb = lBT[buf];
      *(unsigned int*)(bb + (n4 + 0) * 32 + 2 * p) = pack2(bR0[i].x, bR1[i].x);
      *(unsigned int*)(bb + (n4 + 1) * 32 + 2 * p) = pack2(bR0[i].y, bR1[i].y);
      *(unsigned int*)(bb + (n4 + 2) * 32 + 2 * p) = pack2(bR0[i].z, bR1[i].z);
      *(unsigned int*)(bb + (n4 + 3) * 32 + 2 * p) = pack2(bR0[i].w, bR1[i].w);
    }
  };

  fetch(0);
  commit(0);
  for (int ks = 0; ks < ksteps; ks++) {
    const int buf = ks & 1;
    __syncthreads();                         // tile `ks` visible in lA/lBT[buf]
    const bool haveNext = (ks + 1 < ksteps);
    if (haveNext) fetch((ks + 1) * 32);      // global loads in flight over WMMAs
    const v16bf bfr0 = load_fragB(&lBT[buf][(wn + 0) * 32], 32, lane);
    const v16bf bfr1 = load_fragB(&lBT[buf][(wn + 16) * 32], 32, lane);
#pragma unroll
    for (int mf = 0; mf < 4; mf++) {
      const v16bf afr = load_fragA(&lA[buf][(wm + mf * 16) * 32], 32, lane);
      acc[mf][0] = wmma_bf16(afr, bfr0, acc[mf][0]);
      acc[mf][1] = wmma_bf16(afr, bfr1, acc[mf][1]);
    }
    if (haveNext) commit(buf ^ 1);           // other buffer: no barrier needed
  }
  const int half = lane >> 4, lc = lane & 15;
  if (fullN) {
#pragma unroll
    for (int mf = 0; mf < 4; mf++)
#pragma unroll
      for (int nf = 0; nf < 2; nf++)
#pragma unroll
        for (int v = 0; v < 8; v++) {
          const int gm = bm + wm + mf * 16 + half * 8 + v;
          const int gn = bn + wn + nf * 16 + lc;
          const size_t o = (size_t)gm * N + gn;
          float val = acc[mf][nf][v];
          if (epi == 1) val += resid[o];
          C[o] = val;
        }
  } else {
#pragma unroll
    for (int mf = 0; mf < 4; mf++)
#pragma unroll
      for (int nf = 0; nf < 2; nf++)
#pragma unroll
        for (int v = 0; v < 8; v++) {
          const int gm = bm + wm + mf * 16 + half * 8 + v;
          const int gn = bn + wn + nf * 16 + lc;
          if (gm < M && gn < N) {
            const size_t o = (size_t)gm * N + gn;
            float val = acc[mf][nf][v];
            if (epi == 1) val += resid[o];
            C[o] = val;
          }
        }
  }
}

// ---------------------------------------------------------------------------
// Local attention, flash-style. grid = (qchunk 4, window 4, head 16),
// block = 256 (8 waves, 16 q-rows per wave). qkv f32 [2048][3072].
__global__ __launch_bounds__(256) void attn_kernel(
    const float* __restrict__ qkv, const float* __restrict__ biasTab,
    bf16* __restrict__ o) {
  __shared__ __align__(64) bf16 lQ[128 * 64];    // [qrow][d], pre-scaled
  __shared__ __align__(64) bf16 lK[32 * 64];     // [key][d]
  __shared__ __align__(64) bf16 lVT[64 * 32];    // [d][key]
  __shared__ __align__(64) bf16 lP[8][16 * 32];  // per-wave P scratch
  const int head = blockIdx.z, wi = blockIdx.y, qc = blockIdx.x;
  const int tid = threadIdx.x, wid = tid >> 5, lane = tid & 31;
  const int half = lane >> 4, lc = lane & 15;
  const int qbase = qc * 128;
#pragma unroll
  for (int i = 0; i < 16; i++) {           // stage Q (float2), scale = 64^-0.5
    const int idx = tid + i * 256;         // 4096 float2 slots
    const int r = idx >> 5, d2 = (idx & 31) * 2;
    const int n = wi * 512 + qbase + r;
    const float2 qv = *(const float2*)(qkv + (size_t)n * 3072 + head * 64 + d2);
    *(unsigned int*)(lQ + r * 64 + d2) = pack2(qv.x * 0.125f, qv.y * 0.125f);
  }
  __syncthreads();
  const int wrow = wid * 16;
  const v16bf qf0 = load_fragA(&lQ[wrow * 64 + 0], 64, lane);
  const v16bf qf1 = load_fragA(&lQ[wrow * 64 + 32], 64, lane);
  float m[8], l[8];
  v8f oacc[4];
#pragma unroll
  for (int v = 0; v < 8; v++) { m[v] = -1.0e30f; l[v] = 0.0f; }
#pragma unroll
  for (int dn = 0; dn < 4; dn++) oacc[dn] = zero8();

  for (int kc = 0; kc < 32; kc++) {        // 32 key chunks of 32 -> 1024 keys
    __syncthreads();
    if (wi != 0) {                         // uniform: no pads, unguarded loads
#pragma unroll
      for (int i = 0; i < 4; i++) {        // 1024 float2 slots
        const int idx = tid + i * 256;
        const int key = idx >> 5, d2 = (idx & 31) * 2;
        const size_t kg = (size_t)((wi - 1) * 512 + kc * 32 + key);
        const float2 kv = *(const float2*)(qkv + kg * 3072 + 1024 + head * 64 + d2);
        const float2 vv = *(const float2*)(qkv + kg * 3072 + 2048 + head * 64 + d2);
        *(unsigned int*)(lK + key * 64 + d2) = pack2(kv.x, kv.y);
        lVT[(d2 + 0) * 32 + key] = (bf16)vv.x;
        lVT[(d2 + 1) * 32 + key] = (bf16)vv.y;
      }
    } else {                               // window 0: pads -> 0 (masked anyway)
#pragma unroll
      for (int i = 0; i < 4; i++) {
        const int idx = tid + i * 256;
        const int key = idx >> 5, d2 = (idx & 31) * 2;
        const int kg = -512 + kc * 32 + key;
        float2 kv = make_float2(0.0f, 0.0f), vv = make_float2(0.0f, 0.0f);
        if (kg >= 0) {
          kv = *(const float2*)(qkv + (size_t)kg * 3072 + 1024 + head * 64 + d2);
          vv = *(const float2*)(qkv + (size_t)kg * 3072 + 2048 + head * 64 + d2);
        }
        *(unsigned int*)(lK + key * 64 + d2) = pack2(kv.x, kv.y);
        lVT[(d2 + 0) * 32 + key] = (bf16)vv.x;
        lVT[(d2 + 1) * 32 + key] = (bf16)vv.y;
      }
    }
    __syncthreads();
    v8f s0 = zero8(), s1 = zero8();        // sim: 16 rows x 32 keys
    {
      const v16bf b00 = load_fragB(&lK[0 * 64 + 0], 64, lane);
      const v16bf b01 = load_fragB(&lK[0 * 64 + 32], 64, lane);
      s0 = wmma_bf16(qf0, b00, s0);
      s0 = wmma_bf16(qf1, b01, s0);
      const v16bf b10 = load_fragB(&lK[16 * 64 + 0], 64, lane);
      const v16bf b11 = load_fragB(&lK[16 * 64 + 32], 64, lane);
      s1 = wmma_bf16(qf0, b10, s1);
      s1 = wmma_bf16(qf1, b11, s1);
    }
    bf16* Pw = lP[wid];
#pragma unroll
    for (int v = 0; v < 8; v++) {          // bias + mask + online softmax (per row)
      const int iw = qbase + wrow + half * 8 + v;       // 0..511 in window
      const int qg = wi * 512 + iw;
      const int j0 = kc * 32 + lc, j1 = j0 + 16;
      const int kg0 = (wi - 1) * 512 + j0, kg1 = kg0 + 16;
      const bool keep0 = (kg0 >= 0) && (qg >= kg0) && (qg <= kg0 + 512);
      const bool keep1 = (kg1 >= 0) && (qg >= kg1) && (qg <= kg1 + 512);
      int d0 = 512 + iw - j0; d0 = d0 < 0 ? -d0 : d0;
      int d1 = 512 + iw - j1; d1 = d1 < 0 ? -d1 : d1;
      const float x0 = keep0 ? (s0[v] + biasTab[d0 * 16 + head]) : -1.0e30f;
      const float x1 = keep1 ? (s1[v] + biasTab[d1 * 16 + head]) : -1.0e30f;
      float rmax = fmaxf(x0, x1);
#pragma unroll
      for (int sh = 8; sh >= 1; sh >>= 1) rmax = fmaxf(rmax, __shfl_xor(rmax, sh, 32));
      const float newm = fmaxf(m[v], rmax);
      const float corr = __expf(m[v] - newm);
      const float p0 = keep0 ? __expf(x0 - newm) : 0.0f;
      const float p1 = keep1 ? __expf(x1 - newm) : 0.0f;
      float rsum = p0 + p1;
#pragma unroll
      for (int sh = 8; sh >= 1; sh >>= 1) rsum += __shfl_xor(rsum, sh, 32);
      l[v] = l[v] * corr + rsum;
      m[v] = newm;
#pragma unroll
      for (int dn = 0; dn < 4; dn++) oacc[dn][v] *= corr;
      Pw[(half * 8 + v) * 32 + lc] = (bf16)p0;          // D-layout -> A-layout
      Pw[(half * 8 + v) * 32 + 16 + lc] = (bf16)p1;     // bounce through LDS
    }
    const v16bf pa = load_fragA(Pw, 32, lane);           // wave-private, no barrier
#pragma unroll
    for (int dn = 0; dn < 4; dn++) {                     // out += P(16x32) @ V(32x64)
      const v16bf bv = load_fragB(&lVT[(dn * 16) * 32], 32, lane);
      oacc[dn] = wmma_bf16(pa, bv, oacc[dn]);
    }
  }
#pragma unroll
  for (int v = 0; v < 8; v++) {            // o[n][h*64+d] = out / l  (bf16)
    const int n = wi * 512 + qbase + wrow + half * 8 + v;
    const float invl = 1.0f / l[v];
#pragma unroll
    for (int dn = 0; dn < 4; dn++) {
      const int col = head * 64 + dn * 16 + lc;
      o[(size_t)n * 1024 + col] = (bf16)(oacc[dn][v] * invl);
    }
  }
}

// ---------------------------------------------------------------------------
// g[n][f] = bf16( ff[n][f] * gelu(ff[n][2730+f]) ), exact gelu
__global__ __launch_bounds__(256) void swiglu_kernel(
    const float* __restrict__ ff, bf16* __restrict__ g) {
  const int idx = blockIdx.x * 256 + threadIdx.x;
  if (idx >= 2048 * 2730) return;
  const int n = idx / 2730, f = idx - n * 2730;
  const float a = ff[(size_t)n * 5460 + f];
  const float z = ff[(size_t)n * 5460 + 2730 + f];
  const float gl = 0.5f * z * (1.0f + erff(z * 0.70710678118654752f));
  g[idx] = (bf16)(a * gl);
}

// ---------------------------------------------------------------------------
extern "C" void kernel_launch(void* const* d_in, const int* in_sizes, int n_in,
                              void* d_out, int out_size, void* d_ws, size_t ws_size,
                              hipStream_t stream) {
  (void)in_sizes; (void)n_in; (void)out_size; (void)ws_size;
  const int*   tokens  = (const int*)d_in[0];
  const float* tok_emb = (const float*)d_in[1];
  const float* pos_emb = (const float*)d_in[2];
  const float* dpb_w1  = (const float*)d_in[3];
  const float* dpb_b1  = (const float*)d_in[4];
  const float* dpb_w2  = (const float*)d_in[5];
  const float* dpb_b2  = (const float*)d_in[6];
  const float* dpb_w3  = (const float*)d_in[7];
  const float* dpb_b3  = (const float*)d_in[8];
  const float* ln1_g   = (const float*)d_in[9];
  const float* ln1_b   = (const float*)d_in[10];
  const float* Wqkv    = (const float*)d_in[11];
  const float* Wo      = (const float*)d_in[12];
  const float* ln2_g   = (const float*)d_in[13];
  const float* ln2_b   = (const float*)d_in[14];
  const float* Wff1    = (const float*)d_in[15];
  const float* Wff2    = (const float*)d_in[16];
  const float* lnf_g   = (const float*)d_in[17];
  const float* lnf_b   = (const float*)d_in[18];
  const float* Wlogits = (const float*)d_in[19];
  float* out = (float*)d_out;

  char* wsb = (char*)d_ws;
  size_t off = 0;
  float* x       = (float*)(wsb + off); off += (size_t)2048 * 1024 * 4;
  float* qkv     = (float*)(wsb + off); off += (size_t)2048 * 3072 * 4;
  float* ff      = (float*)(wsb + off); off += (size_t)2048 * 5460 * 4;
  float* biasTab = (float*)(wsb + off); off += (size_t)1024 * 16 * 4;
  bf16*  h       = (bf16*)(wsb + off);  off += (size_t)2048 * 1024 * 2;
  bf16*  oattn   = (bf16*)(wsb + off);  off += (size_t)2048 * 1024 * 2;
  bf16*  gbuf    = (bf16*)(wsb + off);  off += (size_t)2048 * 2730 * 2;

  embed_kernel<<<2048, 256, 0, stream>>>(tokens, tok_emb, pos_emb, x);
  dpb_kernel<<<1024, 512, 0, stream>>>(dpb_w1, dpb_b1, dpb_w2, dpb_b2,
                                       dpb_w3, dpb_b3, biasTab);
  for (int layer = 0; layer < 4; layer++) {
    ln_kernel<<<2048, 256, 0, stream>>>(x, ln1_g + layer * 1024, ln1_b + layer * 1024, h);
    gemm_bf16<<<dim3(24, 16), 256, 0, stream>>>(
        h, Wqkv + (size_t)layer * 1024 * 3072, qkv, nullptr, 2048, 3072, 1024, 0);
    attn_kernel<<<dim3(4, 4, 16), 256, 0, stream>>>(qkv, biasTab, oattn);
    gemm_bf16<<<dim3(8, 16), 256, 0, stream>>>(
        oattn, Wo + (size_t)layer * 1024 * 1024, x, x, 2048, 1024, 1024, 1);
    ln_kernel<<<2048, 256, 0, stream>>>(x, ln2_g + layer * 1024, ln2_b + layer * 1024, h);
    gemm_bf16<<<dim3(43, 16), 256, 0, stream>>>(
        h, Wff1 + (size_t)layer * 1024 * 5460, ff, nullptr, 2048, 5460, 1024, 0);
    swiglu_kernel<<<(2048 * 2730 + 255) / 256, 256, 0, stream>>>(ff, gbuf);
    gemm_bf16<<<dim3(8, 16), 256, 0, stream>>>(
        gbuf, Wff2 + (size_t)layer * 2730 * 1024, x, x, 2048, 1024, 2730, 1);
  }
  ln_kernel<<<2048, 256, 0, stream>>>(x, lnf_g, lnf_b, h);
  gemm_bf16<<<dim3(250, 16), 256, 0, stream>>>(
      h, Wlogits, out, nullptr, 2048, 32000, 1024, 0);
}
